// VectorQuantizer_41695542509801
// MI455X (gfx1250) — matrix operand order repaired and checked by reading, not compiled, not run on hardware
//
#include <hip/hip_runtime.h>

typedef float  v8f   __attribute__((ext_vector_type(8)));
typedef __bf16 v16bf __attribute__((ext_vector_type(16)));
typedef __bf16 v8bf  __attribute__((ext_vector_type(8)));
typedef __bf16 bf16_t;
typedef unsigned v4u __attribute__((ext_vector_type(4)));
typedef int      v8i __attribute__((ext_vector_type(8)));
typedef int      v4i __attribute__((ext_vector_type(4)));

#define N_PTS      16384      // 16*32*32 latent vectors
#define DDIM       256
#define VCB        8192
#define VS         4          // V-dimension split across blocks
#define M_TILE     128        // rows per workgroup (8 waves * 16)
#define NT_PER     ((VCB / VS) / 16)   // 128 column-tiles of 16 per block
#define BSTRIDE    264        // 256 + 8 bf16 pad (TDM pad: 4 dwords / row)
#define TILE_BYTES (16 * DDIM * 2)     // 8192 B of bf16 per (hi|lo) tile
#define LOSS_SCALE (1.25f / (float)(N_PTS * DDIM))

union BFPack8  { unsigned short s[8];  float4 f; };
union BFPack16 { unsigned short s[16]; v16bf  v; };

__device__ __forceinline__ unsigned short f2bf_rne(float x) {
  unsigned u = __builtin_bit_cast(unsigned, x);
  return (unsigned short)((u + 0x7fffu + ((u >> 16) & 1u)) >> 16);
}
__device__ __forceinline__ float bf2f(unsigned short b) {
  return __builtin_bit_cast(float, ((unsigned)b) << 16);
}

// ---------------------------------------------------------------------------
// TDM: load one 16-row x 256-bf16 codebook panel into LDS, padding each
// 512B row with 16B (4 dwords) so the LDS stride is 264 bf16 (conflict-free).
// D# fields per CDNA5 ISA (2D tensor, groups 2/3 unused).
// This toolchain exposes the 6-arg builtin: (g0, g1, g2, g3, v8i, cpol).
// ---------------------------------------------------------------------------
__device__ __forceinline__ void tdm_load_tile(unsigned lds_byte_off,
                                              unsigned long long gaddr) {
  v4u g0 = {
      1u,                                            // count=1 (valid user D#)
      lds_byte_off,                                  // LDS destination
      (unsigned)(gaddr & 0xffffffffull),             // global addr [31:0]
      (unsigned)((gaddr >> 32) & 0x01ffffffull) | 0x80000000u  // [56:32]|type=2
  };
  v8i g1 = {
      (int)((1u << 16) |        // data_size = 1 -> 2 bytes
            (1u << 20) |        // pad_enable
            (6u << 22) |        // pad_interval: 2^(6+1)=128 dwords (one row)
            (3u << 25)),        // pad_amount: 3+1 = 4 dwords (16B)
      (int)(256u << 16),        // tensor_dim0 = 256 elements   (bits 79:48)
      (int)(16u << 16),         // tensor_dim1 = 16 rows        (bits 111:80)
      (int)(256u << 16),        // tile_dim0   = 256 elements   (bits 127:112)
      16,                       // tile_dim1   = 16 rows        (bits 143:128)
      256,                      // tensor_dim0_stride = 256     (bits 207:160)
      0, 0                      // tensor_dim1_stride unused (2D)
  };
  v4i z4 = {0, 0, 0, 0};
  v8i z8 = {0, 0, 0, 0, 0, 0, 0, 0};
  __builtin_amdgcn_tensor_load_to_lds(g0, g1, z4, z4, z8, 0);
}

// ---------------------------------------------------------------------------
// Kernel 1: codebook norms (exact f32) + bf16 hi/lo split + zero loss
// ---------------------------------------------------------------------------
__global__ __launch_bounds__(256) void vq_prep_kernel(
    const float* __restrict__ cb, float* __restrict__ cbn,
    bf16_t* __restrict__ cb_hi, bf16_t* __restrict__ cb_lo,
    float* __restrict__ out_loss) {
  const int wave = threadIdx.x >> 5;
  const int lane = threadIdx.x & 31;
  const int row  = blockIdx.x * 8 + wave;
  const size_t off = (size_t)row * DDIM + lane * 8;

  const float4* p = (const float4*)(cb + off);
  float4 a = p[0], b = p[1];
  float x[8] = {a.x, a.y, a.z, a.w, b.x, b.y, b.z, b.w};

  BFPack8 ph, pl;
  float s = 0.0f;
#pragma unroll
  for (int i = 0; i < 8; ++i) {
    s += x[i] * x[i];
    unsigned short h = f2bf_rne(x[i]);
    ph.s[i] = h;
    pl.s[i] = f2bf_rne(x[i] - bf2f(h));
  }
  *(float4*)(cb_hi + off) = ph.f;
  *(float4*)(cb_lo + off) = pl.f;

#pragma unroll
  for (int m = 1; m < 32; m <<= 1) s += __shfl_xor(s, m, 32);
  if (lane == 0) cbn[row] = s;
  if (blockIdx.x == 0 && threadIdx.x == 0) *out_loss = 0.0f;
}

// ---------------------------------------------------------------------------
// Kernel 2: distance GEMM via V_WMMA_F32_16X16X32_BF16 (bf16x3 split),
// TDM double-buffered LDS staging, running argmin.
// grid = (VS, N_PTS / M_TILE), block = 256 (8 wave32)
// ---------------------------------------------------------------------------
__global__ __launch_bounds__(256) void vq_argmin_kernel(
    const float* __restrict__ latents, const bf16_t* __restrict__ cb_hi,
    const bf16_t* __restrict__ cb_lo, const float* __restrict__ cb_norm,
    float* __restrict__ minval_out, int* __restrict__ minidx_out) {
  __shared__ bf16_t sbuf[2][2][16 * BSTRIDE];   // [buffer][hi/lo][tile]

  const int tid  = threadIdx.x;
  const int wave = tid >> 5;
  const int lane = tid & 31;
  const int g    = lane >> 4;      // lane half
  const int nln  = lane & 15;      // A row / B,C column within tile
  const int rowbase = blockIdx.y * M_TILE + wave * 16;
  const int vbase   = blockIdx.x * (VCB / VS);

  // ---- A panel: 16 rows x 256 K, split to bf16 hi/lo in-register -----------
  // 16-bit A layout: lane row = lane%16; elements 0..7 <- K = 32c+8g+0..7,
  //                  elements 8..15 <- K = 32c+16+8g+0..7
  v16bf Ahi[8], Alo[8];
  {
    const float* xrow = latents + (size_t)(rowbase + nln) * DDIM + 8 * g;
#pragma unroll
    for (int c = 0; c < 8; ++c) {
      const float* r0 = xrow + c * 32;
      float x[16];
      *(float4*)&x[0]  = ((const float4*)r0)[0];
      *(float4*)&x[4]  = ((const float4*)r0)[1];
      *(float4*)&x[8]  = ((const float4*)(r0 + 16))[0];
      *(float4*)&x[12] = ((const float4*)(r0 + 16))[1];
      BFPack16 uh, ul;
#pragma unroll
      for (int i = 0; i < 16; ++i) {
        unsigned short h = f2bf_rne(x[i]);
        uh.s[i] = h;
        ul.s[i] = f2bf_rne(x[i] - bf2f(h));
      }
      Ahi[c] = uh.v;
      Alo[c] = ul.v;
    }
  }

  float minv[8];
  int   mini[8];
#pragma unroll
  for (int v = 0; v < 8; ++v) { minv[v] = 3.402823466e38f; mini[v] = 0; }

  // ---- TDM staging setup ---------------------------------------------------
  const unsigned lds_base = (unsigned)(size_t)&sbuf[0][0][0];
  const unsigned lds_half =
      (unsigned)((size_t)&sbuf[0][1][0] - (size_t)&sbuf[0][0][0]);
  const unsigned lds_buf =
      (unsigned)((size_t)&sbuf[1][0][0] - (size_t)&sbuf[0][0][0]);
  const unsigned long long ghi =
      (unsigned long long)(size_t)(cb_hi + (size_t)vbase * DDIM);
  const unsigned long long glo =
      (unsigned long long)(size_t)(cb_lo + (size_t)vbase * DDIM);

  if (wave == 0) {   // prologue: tile 0 -> buffer 0
    tdm_load_tile(lds_base, ghi);
    tdm_load_tile(lds_base + lds_half, glo);
    __builtin_amdgcn_s_wait_tensorcnt(0);
  }
  __syncthreads();

  for (int t = 0; t < NT_PER; ++t) {
    // kick TDM for tile t+1 into the buffer retired at the end of tile t-1
    if (wave == 0 && t + 1 < NT_PER) {
      const unsigned dst = lds_base + (unsigned)((t + 1) & 1) * lds_buf;
      const unsigned long long adv = (unsigned long long)(t + 1) * TILE_BYTES;
      tdm_load_tile(dst, ghi + adv);
      tdm_load_tile(dst + lds_half, glo + adv);
    }

    // ---- compute: 8 K-chunks x 3 split products, 3 accumulator chains ------
    // B layout: lane col = lane%16, elements 0..7 <- K = 32c+8g+0..7, 8..15 <- K = 32c+16+8g+0..7
    const bf16_t* bh = &sbuf[t & 1][0][nln * BSTRIDE + 8 * g];
    const bf16_t* bl = &sbuf[t & 1][1][nln * BSTRIDE + 8 * g];
    v8f a0 = {}, a1 = {}, a2 = {};
#pragma unroll
    for (int c = 0; c < 8; ++c) {
      v8bf h0 = *(const v8bf*)(bh + c * 32);
      v8bf h1 = *(const v8bf*)(bh + c * 32 + 16);
      v8bf l0 = *(const v8bf*)(bl + c * 32);
      v8bf l1 = *(const v8bf*)(bl + c * 32 + 16);
      v16bf Bh = __builtin_shufflevector(h0, h1, 0, 1, 2, 3, 4, 5, 6, 7,
                                         8, 9, 10, 11, 12, 13, 14, 15);
      v16bf Bl = __builtin_shufflevector(l0, l1, 0, 1, 2, 3, 4, 5, 6, 7,
                                         8, 9, 10, 11, 12, 13, 14, 15);
      a0 = __builtin_amdgcn_wmma_f32_16x16x32_bf16(false, Ahi[c], false, Bh,
                                                   (short)0, a0, false, false);
      a1 = __builtin_amdgcn_wmma_f32_16x16x32_bf16(false, Ahi[c], false, Bl,
                                                   (short)0, a1, false, false);
      a2 = __builtin_amdgcn_wmma_f32_16x16x32_bf16(false, Alo[c], false, Bh,
                                                   (short)0, a2, false, false);
    }
    v8f acc = (a0 + a1) + a2;

    const int   vt  = vbase + t * 16;
    const float cbn = cb_norm[vt + nln];
#pragma unroll
    for (int v = 0; v < 8; ++v) {
      float d = cbn - 2.0f * acc[v];   // ||e||^2 - 2 x.e (row const dropped)
      if (d < minv[v]) { minv[v] = d; mini[v] = vt + nln; }
    }

    // publish next buffer: TDM must be complete before anyone reads it
    if (wave == 0 && t + 1 < NT_PER) __builtin_amdgcn_s_wait_tensorcnt(0);
    __syncthreads();
  }

  // ---- cross-lane argmin within each 16-lane half (first-index tiebreak) ---
#pragma unroll
  for (int v = 0; v < 8; ++v) {
    float bv = minv[v];
    int   bi = mini[v];
#pragma unroll
    for (int m = 1; m < 16; m <<= 1) {
      float ov = __shfl_xor(bv, m, 32);
      int   oi = __shfl_xor(bi, m, 32);
      if (ov < bv || (ov == bv && oi < bi)) { bv = ov; bi = oi; }
    }
    if (nln == 0) {
      const int row = rowbase + v + 8 * g;   // C/D layout: M = vgpr + 8*half
      minval_out[blockIdx.x * N_PTS + row] = bv;
      minidx_out[blockIdx.x * N_PTS + row] = bi;
    }
  }
}

// ---------------------------------------------------------------------------
// Kernel 3: merge V-split partials, gather codeword, loss, index output
// ---------------------------------------------------------------------------
__global__ __launch_bounds__(256) void vq_gather_kernel(
    const float* __restrict__ latents, const float* __restrict__ codebook,
    const float* __restrict__ minval, const int* __restrict__ minidx,
    float* __restrict__ out_q, float* __restrict__ out_loss,
    float* __restrict__ out_idx) {
  const int wave = threadIdx.x >> 5;
  const int lane = threadIdx.x & 31;
  const int row  = blockIdx.x * 8 + wave;

  float bv = minval[row];
  int   bi = minidx[row];
#pragma unroll
  for (int s = 1; s < VS; ++s) {
    float v = minval[s * N_PTS + row];
    int   i = minidx[s * N_PTS + row];
    if (v < bv || (v == bv && i < bi)) { bv = v; bi = i; }
  }

  const float4* q4 = (const float4*)(codebook + (size_t)bi * DDIM + lane * 8);
  const float4* x4 = (const float4*)(latents + (size_t)row * DDIM + lane * 8);
  float4*       o4 = (float4*)(out_q + (size_t)row * DDIM + lane * 8);

  float s = 0.0f;
#pragma unroll
  for (int j = 0; j < 2; ++j) {
    float4 q = q4[j];
    float4 x = x4[j];
    o4[j] = q;   // straight-through output == quantized numerically
    float dx = q.x - x.x, dy = q.y - x.y, dz = q.z - x.z, dw = q.w - x.w;
    s += dx * dx + dy * dy + dz * dz + dw * dw;
  }
#pragma unroll
  for (int m = 1; m < 32; m <<= 1) s += __shfl_xor(s, m, 32);
  if (lane == 0) {
    atomicAdd(out_loss, s * LOSS_SCALE);
    out_idx[row] = (float)bi;
  }
}

// ---------------------------------------------------------------------------
extern "C" void kernel_launch(void* const* d_in, const int* in_sizes, int n_in,
                              void* d_out, int out_size, void* d_ws, size_t ws_size,
                              hipStream_t stream) {
  const float* latents  = (const float*)d_in[0];   // [16,32,32,256] f32
  const float* codebook = (const float*)d_in[1];   // [8192,256]     f32

  float* out      = (float*)d_out;
  float* out_q    = out;                        // 4,194,304 floats
  float* out_loss = out + (size_t)N_PTS * DDIM; // 1 float
  float* out_idx  = out_loss + 1;               // 16,384 floats (indices)

  // workspace: cb_norm | cb_hi | cb_lo | minval | minidx  (~8.6 MB)
  float*  cbn    = (float*)d_ws;
  bf16_t* cb_hi  = (bf16_t*)(cbn + VCB);
  bf16_t* cb_lo  = cb_hi + (size_t)VCB * DDIM;
  float*  minval = (float*)(cb_lo + (size_t)VCB * DDIM);
  int*    minidx = (int*)(minval + (size_t)VS * N_PTS);

  vq_prep_kernel<<<VCB / 8, 256, 0, stream>>>(codebook, cbn, cb_hi, cb_lo,
                                              out_loss);

  dim3 gridB(VS, N_PTS / M_TILE);
  vq_argmin_kernel<<<gridB, 256, 0, stream>>>(latents, cb_hi, cb_lo, cbn,
                                              minval, minidx);

  vq_gather_kernel<<<N_PTS / 8, 256, 0, stream>>>(latents, codebook, minval,
                                                  minidx, out_q, out_loss,
                                                  out_idx);
}
